// GraphNeuralNetwork_74732430950951
// MI455X (gfx1250) — matrix operand order repaired and checked by reading, not compiled, not run on hardware
//
#include <hip/hip_runtime.h>
#include <hip/hip_bf16.h>
#include <stdint.h>

typedef _Float16 f16;
typedef __attribute__((ext_vector_type(16))) _Float16 v16h;
typedef __attribute__((ext_vector_type(8)))  _Float16 v8h;
typedef __attribute__((ext_vector_type(8)))  float    v8f;
typedef long long i64;

#define NNODES 100000
#define NEDGES 1600000
#define NGRAPH 16

// ---------------- small utility kernels ----------------

__global__ void k_fill_f32(float* p, float v, int n) {
    int i = blockIdx.x * blockDim.x + threadIdx.x;
    if (i < n) p[i] = v;
}

__global__ void k_deg_edges(float* deg, const i64* __restrict__ dst,
                            const float* __restrict__ w, int E) {
    int e = blockIdx.x * blockDim.x + threadIdx.x;
    if (e < E) atomicAdd(&deg[(int)dst[e]], w[e]);
}

__global__ void k_dinv(float* dinv, const float* __restrict__ deg, int n) {
    int i = blockIdx.x * blockDim.x + threadIdx.x;
    if (i < n) { float d = deg[i]; dinv[i] = (d > 0.f) ? rsqrtf(d) : 0.f; }
}

__global__ void k_norm(float* norm, const i64* __restrict__ src, const i64* __restrict__ dst,
                       const float* __restrict__ w, const float* __restrict__ dinv, int E) {
    int e = blockIdx.x * blockDim.x + threadIdx.x;
    if (e < E) norm[e] = dinv[(int)src[e]] * w[e] * dinv[(int)dst[e]];
}

// f32 [R,K] row-major -> f16 [RPad,K], zero-pad extra rows
__global__ void k_cvt_pad_f16(f16* __restrict__ out, const float* __restrict__ in,
                              int R, int RPad, int K) {
    int idx = blockIdx.x * blockDim.x + threadIdx.x;
    if (idx >= RPad * K) return;
    int r = idx / K;
    f16 v = (f16)0.f;
    if (r < R) v = (f16)in[idx];
    out[idx] = v;
}

// ---------------- WMMA GEMM: out[M,nAct](f16) = A[M,K](f16) x B[nPad,K](f16)^T ----------------
// block = 256 threads = 8 waves laid out 4(M) x 2(N); block tile 64x32; wave tile 16x16; K-step 32.
__global__ __launch_bounds__(256)
void k_gemm_wmma(const f16* __restrict__ A, const f16* __restrict__ B,
                 const float* __restrict__ bias, f16* __restrict__ out,
                 int M, int K, int nAct, int relu) {
    const int lane = threadIdx.x & 31;
    const int wave = threadIdx.x >> 5;
    const int ll   = lane & 15;
    const int hi   = lane >> 4;            // 0: lanes 0-15, 1: lanes 16-31
    const int m_base = blockIdx.x * 64 + (wave & 3) * 16;
    const int n_base = blockIdx.y * 32 + (wave >> 2) * 16;

    int arow = m_base + ll; if (arow >= M) arow = M - 1;   // clamp: keep EXEC full for WMMA
    const f16* pa = A + (size_t)arow * K;
    const f16* pb = B + (size_t)(n_base + ll) * K;         // B rows padded -> always valid

    v8f c = {};
    for (int kt = 0; kt < K; kt += 32) {
        // A 16x32 layout: lanes 0-15 hold K {0..7,16..23}, lanes 16-31 hold K {8..15,24..31}
        v8h a0 = *(const v8h*)(pa + kt + hi * 8);
        v8h a1 = *(const v8h*)(pa + kt + 16 + hi * 8);
        // B 32x16 layout: lanes 0-15 hold K {0..7,16..23}, lanes 16-31 hold K {8..15,24..31}
        v8h b0 = *(const v8h*)(pb + kt + hi * 8);
        v8h b1 = *(const v8h*)(pb + kt + 16 + hi * 8);
        v16h a, b;
#pragma unroll
        for (int i = 0; i < 8; ++i) { a[i] = a0[i]; a[i + 8] = a1[i]; b[i] = b0[i]; b[i + 8] = b1[i]; }
        c = __builtin_amdgcn_wmma_f32_16x16x32_f16(false, a, false, b, (short)0, c, false, false);
    }

    const int n = n_base + ll;
    if (n < nAct) {
        float bv = bias ? bias[n] : 0.f;
#pragma unroll
        for (int v = 0; v < 8; ++v) {
            int m = m_base + v + 8 * hi;   // C/D layout: VGPR v -> row v (lanes 0-15) / v+8 (16-31)
            if (m < M) {
                float val = c[v] + bv;
                if (relu) val = fmaxf(val, 0.f);
                out[(size_t)m * nAct + n] = (f16)val;
            }
        }
    }
}

// ---------------- GCN aggregation ----------------

// agg[i,f] = H[i,f] * dinv[i]^2   (self-loop term; also initializes agg)
__global__ void k_selfloop(float* __restrict__ agg, const f16* __restrict__ H,
                           const float* __restrict__ dinv, int cout) {
    int i = blockIdx.x, f = threadIdx.x;
    float dv = dinv[i];
    agg[(size_t)i * cout + f] = (float)H[(size_t)i * cout + f] * dv * dv;
}

// one block per edge, one thread per feature
__global__ void k_scatter(float* __restrict__ agg, const f16* __restrict__ H,
                          const i64* __restrict__ src, const i64* __restrict__ dst,
                          const float* __restrict__ norm, int cout) {
    int e = blockIdx.x, f = threadIdx.x;
    int s = (int)src[e], d = (int)dst[e];
    float nm = norm[e];
    atomicAdd(&agg[(size_t)d * cout + f], (float)H[(size_t)s * cout + f] * nm);
}

__global__ void k_bias_relu(f16* __restrict__ out, const float* __restrict__ agg,
                            const float* __restrict__ bias, int cout) {
    int i = blockIdx.x, f = threadIdx.x;
    float v = agg[(size_t)i * cout + f] + bias[f];
    out[(size_t)i * cout + f] = (f16)fmaxf(v, 0.f);
}

// ---------------- pooling + head ----------------

__global__ void k_pool_cnt(float* cnt, const i64* __restrict__ batch, int n) {
    int i = blockIdx.x * blockDim.x + threadIdx.x;
    if (i < n) atomicAdd(&cnt[(int)batch[i]], 1.f);
}

__global__ void k_pool_sum(float* __restrict__ sums, const f16* __restrict__ X,
                           const i64* __restrict__ batch, int cdim) {
    int i = blockIdx.x, f = threadIdx.x;
    if (f < cdim) atomicAdd(&sums[(int)batch[i] * cdim + f], (float)X[(size_t)i * cdim + f]);
}

// 16 waves, one graph per wave: sigmoid(mean-pool . W2 + b2)
__global__ void k_final(float* __restrict__ out, const float* __restrict__ sums,
                        const float* __restrict__ cnt, const float* __restrict__ W2,
                        const float* __restrict__ b2, int cdim) {
    int g = threadIdx.x >> 5;
    int lane = threadIdx.x & 31;
    float c = fmaxf(cnt[g], 1.f);
    float acc = 0.f;
    for (int f = lane; f < cdim; f += 32) acc += (sums[g * cdim + f] / c) * W2[f];
#pragma unroll
    for (int off = 16; off; off >>= 1) acc += __shfl_down(acc, off, 32);
    if (lane == 0) out[g] = 1.f / (1.f + __expf(-(acc + b2[0])));
}

// ---------------- launcher ----------------

extern "C" void kernel_launch(void* const* d_in, const int* in_sizes, int n_in,
                              void* d_out, int out_size, void* d_ws, size_t ws_size,
                              hipStream_t stream) {
    (void)in_sizes; (void)n_in; (void)out_size; (void)ws_size;
    const float* x    = (const float*)d_in[0];
    const i64*   eidx = (const i64*)d_in[1];            // [2, E]: src row then dst row
    const float* ew   = (const float*)d_in[2];
    const i64*   batch= (const i64*)d_in[3];
    const float* Wc[3] = {(const float*)d_in[4], (const float*)d_in[6], (const float*)d_in[8]};
    const float* bc[3] = {(const float*)d_in[5], (const float*)d_in[7], (const float*)d_in[9]};
    const float* Wl  = (const float*)d_in[10]; const float* bl  = (const float*)d_in[11];
    const float* Wm0 = (const float*)d_in[12]; const float* bm0 = (const float*)d_in[13];
    const float* Wm1 = (const float*)d_in[14]; const float* bm1 = (const float*)d_in[15];
    const float* W2  = (const float*)d_in[16]; const float* b2  = (const float*)d_in[17];
    float* out = (float*)d_out;

    const i64* src = eidx;
    const i64* dst = eidx + NEDGES;

    // workspace carving
    char* ws = (char*)d_ws;
    size_t off = 0;
    auto carve = [&](size_t bytes) -> void* {
        void* p = ws + off;
        off = (off + bytes + 255) & ~(size_t)255;
        return p;
    };
    float* deg  = (float*)carve((size_t)NNODES * 4);
    float* dinv = (float*)carve((size_t)NNODES * 4);
    float* norm = (float*)carve((size_t)NEDGES * 4);
    // padded f16 weights: rows padded to mult of 32 for the 32-wide N block tile
    const int wR[6]    = {64, 96, 128, 512, 256, 170};   // actual rows (cout)
    const int wRP[6]   = {64, 96, 128, 512, 256, 192};   // padded rows
    const int wK[6]    = {32, 64, 96, 128, 512, 256};    // cols (cin)
    const float* wSrc[6] = {Wc[0], Wc[1], Wc[2], Wl, Wm0, Wm1};
    f16* Wh[6];
    for (int i = 0; i < 6; ++i) Wh[i] = (f16*)carve((size_t)wRP[i] * wK[i] * 2);
    f16*  Xh0 = (f16*)carve((size_t)NNODES * 512 * 2);   // holds widths 32,96,512,170
    f16*  Xh1 = (f16*)carve((size_t)NNODES * 256 * 2);   // holds widths 64,128,256
    f16*  Hh  = (f16*)carve((size_t)NNODES * 128 * 2);
    float* agg = (float*)carve((size_t)NNODES * 128 * 4);
    float* sums = (float*)carve((size_t)NGRAPH * 170 * 4);
    float* cnt  = (float*)carve((size_t)NGRAPH * 4);

    const int T = 256;
    auto cdiv = [](int a, int b) { return (a + b - 1) / b; };

    // degree / normalization (deg starts at 1.0 from the self loop)
    k_fill_f32<<<cdiv(NNODES, T), T, 0, stream>>>(deg, 1.0f, NNODES);
    k_deg_edges<<<cdiv(NEDGES, T), T, 0, stream>>>(deg, dst, ew, NEDGES);
    k_dinv<<<cdiv(NNODES, T), T, 0, stream>>>(dinv, deg, NNODES);
    k_norm<<<cdiv(NEDGES, T), T, 0, stream>>>(norm, src, dst, ew, dinv, NEDGES);

    // weight + input conversion to f16
    for (int i = 0; i < 6; ++i) {
        int tot = wRP[i] * wK[i];
        k_cvt_pad_f16<<<cdiv(tot, T), T, 0, stream>>>(Wh[i], wSrc[i], wR[i], wRP[i], wK[i]);
    }
    k_cvt_pad_f16<<<cdiv(NNODES * 32, T), T, 0, stream>>>(Xh0, x, NNODES, NNODES, 32);

    // 3 GCNConv layers: GEMM -> selfloop init -> edge scatter -> bias+relu
    f16* cur = Xh0; f16* nxt = Xh1;
    for (int L = 0; L < 3; ++L) {
        int cin = wK[L], cout = wR[L];
        dim3 gg(cdiv(NNODES, 64), wRP[L] / 32);
        k_gemm_wmma<<<gg, 256, 0, stream>>>(cur, Wh[L], nullptr, Hh, NNODES, cin, cout, 0);
        k_selfloop<<<NNODES, cout, 0, stream>>>(agg, Hh, dinv, cout);
        k_scatter<<<NEDGES, cout, 0, stream>>>(agg, Hh, src, dst, norm, cout);
        k_bias_relu<<<NNODES, cout, 0, stream>>>(nxt, agg, bc[L], cout);
        f16* t = cur; cur = nxt; nxt = t;
    }
    // after loop: cur = Xh1 (128-wide activations)

    // MLP: 128->512->256->170 with fused bias+relu epilogues
    const float* mb[3] = {bl, bm0, bm1};
    f16* mOut[3] = {Xh0, Xh1, Xh0};
    f16* mIn = cur;
    for (int L = 0; L < 3; ++L) {
        int wi = 3 + L;
        dim3 gg(cdiv(NNODES, 64), wRP[wi] / 32);
        k_gemm_wmma<<<gg, 256, 0, stream>>>(mIn, Wh[wi], mb[L], mOut[L], NNODES, wK[wi], wR[wi], 1);
        mIn = mOut[L];
    }
    // mIn = Xh0, 170-wide

    // mean pool + head
    k_fill_f32<<<cdiv(NGRAPH * 170, T), T, 0, stream>>>(sums, 0.f, NGRAPH * 170);
    k_fill_f32<<<1, NGRAPH, 0, stream>>>(cnt, 0.f, NGRAPH);
    k_pool_cnt<<<cdiv(NNODES, T), T, 0, stream>>>(cnt, batch, NNODES);
    k_pool_sum<<<NNODES, 192, 0, stream>>>(sums, mIn, batch, 170);
    k_final<<<1, 512, 0, stream>>>(out, sums, cnt, W2, b2, 170);
}
